// NewAttention_39642548142828
// MI455X (gfx1250) — matrix operand order repaired and checked by reading, not compiled, not run on hardware
//
#include <hip/hip_runtime.h>
#include <hip/hip_bf16.h>

// ---------------------------------------------------------------------------
// Problem constants (match reference: B,S,H,NH,HD = 8,1024,1024,16,64; VALID=512)
// ---------------------------------------------------------------------------
#define T_TOK  4096   // B * VALID
#define HID    1024
#define NHEAD  16
#define HDIM   64
#define BATCH  8
#define VALIDT 512

typedef __bf16 bf16_t;
typedef __attribute__((ext_vector_type(16))) __bf16 v16bf;
typedef __attribute__((ext_vector_type(8)))  float  v8f;

// ---------------------------------------------------------------------------
// Async global->LDS copy (CDNA5 GLOBAL_LOAD_ASYNC_TO_LDS_B128, ASYNCcnt).
// Builtin signature (from compiler diagnostic):
//   void __builtin_amdgcn_global_load_async_to_lds_b128(
//       int __vector(4) __device__ *gsrc, int __vector(4) __shared__ *ldst,
//       imm int offset, imm int cpol)
// ---------------------------------------------------------------------------
#if defined(__has_builtin)
#if __has_builtin(__builtin_amdgcn_global_load_async_to_lds_b128) && \
    __has_builtin(__builtin_amdgcn_s_wait_asynccnt)
#define HAVE_ASYNC_LDS 1
#endif
#endif
#ifndef HAVE_ASYNC_LDS
#define HAVE_ASYNC_LDS 0
#endif

#if HAVE_ASYNC_LDS
typedef int v4i_gcc __attribute__((vector_size(16)));
typedef __attribute__((address_space(1))) v4i_gcc gv4i;   // global (device) AS
typedef __attribute__((address_space(3))) v4i_gcc lv4i;   // LDS (shared) AS
#endif

__device__ __forceinline__ void cp16_async(void* lds_dst, const void* gsrc) {
#if HAVE_ASYNC_LDS
  __builtin_amdgcn_global_load_async_to_lds_b128(
      (gv4i*)gsrc, (lv4i*)lds_dst, /*offset=*/0, /*cpol=*/0);
#else
  *reinterpret_cast<uint4*>(lds_dst) = *reinterpret_cast<const uint4*>(gsrc);
#endif
}

__device__ __forceinline__ void async_wait() {
#if HAVE_ASYNC_LDS
  __builtin_amdgcn_s_wait_asynccnt(0);
#endif
}

// ---------------------------------------------------------------------------
// WMMA helper: D = A(16x32 bf16) * B(32x16 bf16) + C(16x16 f32)
// ---------------------------------------------------------------------------
__device__ __forceinline__ v8f wmma_bf16(v16bf a, v16bf b, v8f c) {
  return __builtin_amdgcn_wmma_f32_16x16x32_bf16(
      /*neg_a=*/false, a, /*neg_b=*/false, b,
      /*c_mod=*/(short)0, c, /*reuse_a=*/false, /*reuse_b=*/false);
}

// ---------------------------------------------------------------------------
// Fragment loaders from LDS (wave32; contiguous k = 16h + s order, same for A and B).
// Both loaders read two contiguous 16B chunks per lane -> ds_load_b128 pairs.
//
// A 16x32 bf16: lane l (m=l&15, h=l>>4); VGPR j holds K = h*16+2j, 2j+1
// ---------------------------------------------------------------------------
__device__ __forceinline__ v16bf frag_a_lds(const bf16_t* base, int stride) {
  const int l = threadIdx.x & 31;
  const int m = l & 15, h = l >> 4;
  const bf16_t* row = base + m * stride;
  v16bf f;
#pragma unroll
  for (int j = 0; j < 8; ++j) {
    const int kb = h * 16 + j * 2; // h*16+2j
    f[2 * j]     = row[kb];
    f[2 * j + 1] = row[kb + 1];
  }
  return f;
}

// B 32x16 where LDS holds B^T row-major (N x K): lane l (n=l&15, h=l>>4);
// VGPR j holds K = h*16 + 2j, 2j+1 -> one contiguous 32B run per lane.
__device__ __forceinline__ v16bf frag_bt_lds(const bf16_t* base, int stride) {
  const int l = threadIdx.x & 31;
  const int n = l & 15, h = l >> 4;
  const bf16_t* row = base + n * stride + h * 16;
  v16bf f;
#pragma unroll
  for (int j = 0; j < 8; ++j) {
    f[2 * j]     = row[2 * j];
    f[2 * j + 1] = row[2 * j + 1];
  }
  return f;
}

// ---------------------------------------------------------------------------
// Kernel: f32 -> bf16 convert (grid-stride)
// ---------------------------------------------------------------------------
__global__ __launch_bounds__(256)
void cvt_f32_bf16(const float* __restrict__ src, bf16_t* __restrict__ dst, int n) {
  int i = blockIdx.x * blockDim.x + threadIdx.x;
  const int stride = gridDim.x * blockDim.x;
  for (; i < n; i += stride) dst[i] = (bf16_t)src[i];
}

// ---------------------------------------------------------------------------
// Kernel: f32 (K x N) -> bf16 transposed (N x K), LDS-tiled 64x64.
// Coalesced f32 reads, coalesced bf16 writes.
// ---------------------------------------------------------------------------
__global__ __launch_bounds__(256)
void cvt_transpose_kernel(const float* __restrict__ src, bf16_t* __restrict__ dst,
                          int K, int N) {
  __shared__ bf16_t tile[64][72];
  const int k0 = blockIdx.y * 64, n0 = blockIdx.x * 64;
#pragma unroll
  for (int i = 0; i < 16; ++i) {
    const int u = threadIdx.x + i * 256;
    const int r = u >> 6, c = u & 63;
    tile[r][c] = (bf16_t)src[(size_t)(k0 + r) * N + n0 + c];
  }
  __syncthreads();
#pragma unroll
  for (int i = 0; i < 16; ++i) {
    const int u = threadIdx.x + i * 256;
    const int nn = u >> 6, kk = u & 63;
    dst[(size_t)(n0 + nn) * K + k0 + kk] = tile[kk][nn];
  }
}

// ---------------------------------------------------------------------------
// Kernel: bf16 GEMM  C[MxN] = A[MxK] * Bt[NxK]^T + bias[N]
// B pre-transposed (N x K): both tiles stage identically, every fragment load
// is contiguous (ds_load_b128). Double-buffered async global->LDS pipeline.
// 256 threads = 8 waves (4 M-waves x 2 N-waves); WG tile 128x128, wave 32x64.
// ---------------------------------------------------------------------------
template <bool OUT_BF16>
__global__ __launch_bounds__(256)
void gemm_bf16_kernel(const bf16_t* __restrict__ A, const bf16_t* __restrict__ Bt,
                      const float* __restrict__ bias, void* __restrict__ Cout,
                      int M, int N, int K) {
  __shared__ __align__(16) bf16_t As[2][128][40];    // 80B rows (16B aligned)
  __shared__ __align__(16) bf16_t Bts[2][128][40];   // B^T tiles: 128 n x 32 k

  const int tid = threadIdx.x;
  const int wid = tid >> 5;
  const int wm = wid & 3, wn = wid >> 2;
  const int mBase = blockIdx.y * 128, nBase = blockIdx.x * 128;

  auto stage = [&](int buf, int k0) {
#pragma unroll
    for (int it = 0; it < 2; ++it) {
      const int u = tid + it * 256;
      const int row = u >> 2, q = u & 3;
      cp16_async(&As[buf][row][q * 8],
                 A + (size_t)(mBase + row) * K + k0 + q * 8);
      cp16_async(&Bts[buf][row][q * 8],
                 Bt + (size_t)(nBase + row) * K + k0 + q * 8);
    }
  };

  const v8f vzero = {0.f, 0.f, 0.f, 0.f, 0.f, 0.f, 0.f, 0.f};
  v8f acc[2][4];
#pragma unroll
  for (int i = 0; i < 2; ++i)
#pragma unroll
    for (int j = 0; j < 4; ++j) acc[i][j] = vzero;

  stage(0, 0);
  async_wait();
  __syncthreads();

  const int nk = K >> 5;
  for (int kt = 0; kt < nk; ++kt) {
    const int cur = kt & 1;
    if (kt + 1 < nk) stage(cur ^ 1, (kt + 1) * 32);   // overlap fetch w/ WMMA
    if (kt + 2 < nk) {                                 // warm L2 two tiles ahead
      __builtin_prefetch(A + (size_t)(mBase + (tid >> 1)) * K + (kt + 2) * 32, 0, 3);
      __builtin_prefetch(Bt + (size_t)(nBase + (tid >> 1)) * K + (kt + 2) * 32, 0, 3);
    }

    v16bf am[2];
    am[0] = frag_a_lds(&As[cur][wm * 32][0], 40);
    am[1] = frag_a_lds(&As[cur][wm * 32 + 16][0], 40);
    v16bf bn[4];
#pragma unroll
    for (int j = 0; j < 4; ++j)
      bn[j] = frag_bt_lds(&Bts[cur][wn * 64 + j * 16][0], 40);
#pragma unroll
    for (int i = 0; i < 2; ++i)
#pragma unroll
      for (int j = 0; j < 4; ++j) acc[i][j] = wmma_bf16(am[i], bn[j], acc[i][j]);

    async_wait();
    __syncthreads();
  }

  // epilogue: C/D layout -> lane l holds rows r+8*(l>>4), col l&15
  const int l = tid & 31, h = l >> 4, n = l & 15;
#pragma unroll
  for (int i = 0; i < 2; ++i)
#pragma unroll
    for (int j = 0; j < 4; ++j) {
      const int col = nBase + wn * 64 + j * 16 + n;
      const float bv = bias ? bias[col] : 0.0f;
#pragma unroll
      for (int r = 0; r < 8; ++r) {
        const int row = mBase + wm * 32 + i * 16 + h * 8 + r;
        const float v = acc[i][j][r] + bv;
        if (OUT_BF16)
          reinterpret_cast<bf16_t*>(Cout)[(size_t)row * N + col] = (bf16_t)v;
        else
          reinterpret_cast<float*>(Cout)[(size_t)row * N + col] = v;
      }
    }
}

// ---------------------------------------------------------------------------
// Kernel: RoPE + split qkv(T x 3H, bf16):
//   q -> [(B*NH) x VALID x HD], PRE-SCALED by 1/sqrt(HD) (folds attn scale)
//   k -> [(B*NH) x VALID x HD]
//   v -> [(B*NH) x HD x VALID] transposed (so P*V B-fragments are contiguous)
// ---------------------------------------------------------------------------
__global__ __launch_bounds__(256)
void rope_split_kernel(const bf16_t* __restrict__ qkv,
                       const float* __restrict__ cosp, const float* __restrict__ sinp,
                       bf16_t* __restrict__ qb, bf16_t* __restrict__ kb,
                       bf16_t* __restrict__ vt) {
  const int idx = blockIdx.x * blockDim.x + threadIdx.x;
  if (idx >= T_TOK * HID) return;
  const int t = idx >> 10;          // / HID
  const int c = idx & (HID - 1);
  const int hh = c >> 6, d = c & 63;
  const size_t base = (size_t)t * (3 * HID);

  const float qv = (float)qkv[base + c];
  const float kv = (float)qkv[base + HID + c];
  const float vv = (float)qkv[base + 2 * HID + c];
  const int d2 = (d < 32) ? d + 32 : d - 32;
  const float sgn = (d < 32) ? -1.0f : 1.0f;
  const float qp = (float)qkv[base + hh * 64 + d2];
  const float kp = (float)qkv[base + HID + hh * 64 + d2];
  const float co = cosp[t * HDIM + d];
  const float si = sinp[t * HDIM + d];

  const float qo = (qv * co + sgn * qp * si) * 0.125f;  // 1/sqrt(HD) folded in
  const float ko = kv * co + sgn * kp * si;

  const int b = t / VALIDT, row = t % VALIDT;
  const int bh = b * NHEAD + hh;
  const size_t o = ((size_t)bh * VALIDT + row) * HDIM + d;
  qb[o] = (bf16_t)qo;
  kb[o] = (bf16_t)ko;
  vt[((size_t)bh * HDIM + d) * VALIDT + row] = (bf16_t)vv;
}

// ---------------------------------------------------------------------------
// Kernel: flash attention over valid tokens only.
// Grid: (B*NH) * (VALID/64) workgroups; 128 threads = 4 waves, 16 queries/wave.
// S = Q*K^T (bf16 WMMA, f32 acc, Q pre-scaled), online softmax (half-group
// shfl reductions), P via per-wave LDS re-fragmentation, ctx += P*V (WMMA).
// K/V tiles double-buffered through the async global->LDS path.
// ---------------------------------------------------------------------------
__global__ __launch_bounds__(128)
void attn_kernel(const bf16_t* __restrict__ qb, const bf16_t* __restrict__ kb,
                 const bf16_t* __restrict__ vt, bf16_t* __restrict__ ctx) {
  __shared__ __align__(16) bf16_t Qs[64][72];
  __shared__ __align__(16) bf16_t Ks[2][64][72];   // keys x hd  (B^T for Q*K^T)
  __shared__ __align__(16) bf16_t Vts[2][64][72];  // hd x keys  (B^T for P*V)
  __shared__ __align__(16) bf16_t Ps[4][16][72];

  const int tid = threadIdx.x;
  const int wid = tid >> 5;
  const int l = tid & 31, h = l >> 4, n = l & 15;
  const int qblk = blockIdx.x & 7;          // 64-query block within sequence
  const int bh = blockIdx.x >> 3;           // b*NH + head
  const size_t headBase = (size_t)bh * VALIDT * HDIM;

  auto stageKV = [&](int buf, int kb0) {
#pragma unroll
    for (int it = 0; it < 4; ++it) {
      const int u = tid + it * 128;
      const int row = u >> 3, q = u & 7;
      cp16_async(&Ks[buf][row][q * 8],
                 kb + headBase + (size_t)(kb0 + row) * HDIM + q * 8);
      cp16_async(&Vts[buf][row][q * 8],
                 vt + headBase + (size_t)row * VALIDT + kb0 + q * 8);
    }
  };

  // stage Q tile (64 x 64) and first K/V block
#pragma unroll
  for (int it = 0; it < 4; ++it) {
    const int u = tid + it * 128;
    const int row = u >> 3, q = u & 7;
    cp16_async(&Qs[row][q * 8],
               qb + headBase + (size_t)(qblk * 64 + row) * HDIM + q * 8);
  }
  stageKV(0, 0);
  async_wait();
  __syncthreads();

  const v16bf aq0 = frag_a_lds(&Qs[wid * 16][0], 72);
  const v16bf aq1 = frag_a_lds(&Qs[wid * 16][32], 72);

  const v8f vzero = {0.f, 0.f, 0.f, 0.f, 0.f, 0.f, 0.f, 0.f};
  v8f cacc[4];
  float m_run[8], l_run[8];
#pragma unroll
  for (int j = 0; j < 4; ++j) cacc[j] = vzero;
#pragma unroll
  for (int r = 0; r < 8; ++r) { m_run[r] = -1e30f; l_run[r] = 0.f; }

  const int nkb = VALIDT / 64;
  for (int kblk = 0; kblk < nkb; ++kblk) {
    const int cur = kblk & 1;
    if (kblk + 1 < nkb) stageKV(cur ^ 1, (kblk + 1) * 64);  // overlap w/ WMMA

    // scores: 4 n-tiles x (HD=64 -> 2 k-steps) WMMA; K tile row-major == B^T
    v8f s[4];
#pragma unroll
    for (int nt = 0; nt < 4; ++nt) {
      const v16bf bk0 = frag_bt_lds(&Ks[cur][nt * 16][0], 72);
      const v16bf bk1 = frag_bt_lds(&Ks[cur][nt * 16][32], 72);
      s[nt] = wmma_bf16(aq0, bk0, vzero);
      s[nt] = wmma_bf16(aq1, bk1, s[nt]);
    }

    // online softmax; each C-layout row lives in the 16 lanes of one half-wave
    float p[4][8];
#pragma unroll
    for (int r = 0; r < 8; ++r) {
      float mx = s[0][r];
#pragma unroll
      for (int nt = 1; nt < 4; ++nt) mx = fmaxf(mx, s[nt][r]);
      mx = fmaxf(mx, __shfl_xor(mx, 1, 32));
      mx = fmaxf(mx, __shfl_xor(mx, 2, 32));
      mx = fmaxf(mx, __shfl_xor(mx, 4, 32));
      mx = fmaxf(mx, __shfl_xor(mx, 8, 32));
      const float m_new = fmaxf(m_run[r], mx);
      const float corr = __expf(m_run[r] - m_new);
      m_run[r] = m_new;
      float rs = 0.f;
#pragma unroll
      for (int nt = 0; nt < 4; ++nt) {
        const float pv = __expf(s[nt][r] - m_new);
        p[nt][r] = pv;
        rs += pv;
      }
      rs += __shfl_xor(rs, 1, 32);
      rs += __shfl_xor(rs, 2, 32);
      rs += __shfl_xor(rs, 4, 32);
      rs += __shfl_xor(rs, 8, 32);
      l_run[r] = l_run[r] * corr + rs;
#pragma unroll
      for (int j = 0; j < 4; ++j) cacc[j][r] *= corr;
    }

    // re-fragment P through per-wave LDS (C-layout -> A-operand layout)
#pragma unroll
    for (int nt = 0; nt < 4; ++nt)
#pragma unroll
      for (int r = 0; r < 8; ++r)
        Ps[wid][h * 8 + r][nt * 16 + n] = (bf16_t)p[nt][r];

    const v16bf ap0 = frag_a_lds(&Ps[wid][0][0], 72);
    const v16bf ap1 = frag_a_lds(&Ps[wid][0][32], 72);
#pragma unroll
    for (int j = 0; j < 4; ++j) {
      // B[k][n] = V[key k][hd n] = Vts[n][k]  (contiguous along k)
      const v16bf bv0 = frag_bt_lds(&Vts[cur][j * 16][0], 72);
      const v16bf bv1 = frag_bt_lds(&Vts[cur][j * 16][32], 72);
      cacc[j] = wmma_bf16(ap0, bv0, cacc[j]);
      cacc[j] = wmma_bf16(ap1, bv1, cacc[j]);
    }

    async_wait();
    __syncthreads();
  }

  // epilogue: normalize, write ctx (T x HID) bf16, merged-head layout
  const int hh = bh & (NHEAD - 1);
  const int b = bh >> 4;
#pragma unroll
  for (int r = 0; r < 8; ++r) {
    const int qrow = qblk * 64 + wid * 16 + h * 8 + r;
    const int t = b * VALIDT + qrow;
    const float inv = 1.0f / l_run[r];
#pragma unroll
    for (int j = 0; j < 4; ++j)
      ctx[(size_t)t * HID + hh * 64 + j * 16 + n] = (bf16_t)(cacc[j][r] * inv);
  }
}

// ---------------------------------------------------------------------------
// Host launcher
// Inputs: 0 hidden_states(T,H) f32 | 1 cos(T,1,64) | 2 sin(T,1,64) |
//         3 attention_bias | 4 qkv_w(H,3H) | 5 qkv_b(3H) | 6 o_w(H,H) |
//         7 o_b(H) | 8 indices | 9 batch | 10 seqlen
// Output: (T, H) f32
// ---------------------------------------------------------------------------
extern "C" void kernel_launch(void* const* d_in, const int* in_sizes, int n_in,
                              void* d_out, int out_size, void* d_ws, size_t ws_size,
                              hipStream_t stream) {
  (void)in_sizes; (void)n_in; (void)out_size; (void)ws_size;
  const float* hs    = (const float*)d_in[0];
  const float* cosp  = (const float*)d_in[1];
  const float* sinp  = (const float*)d_in[2];
  const float* qkv_w = (const float*)d_in[4];
  const float* qkv_b = (const float*)d_in[5];
  const float* o_w   = (const float*)d_in[6];
  const float* o_b   = (const float*)d_in[7];

  char* ws = (char*)d_ws;
  size_t off = 0;
  bf16_t* hs_b   = (bf16_t*)(ws + off); off += (size_t)T_TOK * HID * 2;
  bf16_t* qkvw_t = (bf16_t*)(ws + off); off += (size_t)HID * 3 * HID * 2; // (3H x H)
  bf16_t* ow_t   = (bf16_t*)(ws + off); off += (size_t)HID * HID * 2;     // (H x H)
  bf16_t* qkv    = (bf16_t*)(ws + off); off += (size_t)T_TOK * 3 * HID * 2;
  bf16_t* qbuf   = (bf16_t*)(ws + off); off += (size_t)T_TOK * HID * 2;
  bf16_t* kbuf   = (bf16_t*)(ws + off); off += (size_t)T_TOK * HID * 2;
  bf16_t* vtbuf  = (bf16_t*)(ws + off); off += (size_t)T_TOK * HID * 2;
  bf16_t* ctxb   = qkv;  // qkv region dead after rope_split -> reuse for ctx

  cvt_f32_bf16<<<512, 256, 0, stream>>>(hs, hs_b, T_TOK * HID);
  // weights converted AND transposed: (K x N) f32 -> (N x K) bf16
  cvt_transpose_kernel<<<dim3(3 * HID / 64, HID / 64), 256, 0, stream>>>(
      qkv_w, qkvw_t, HID, 3 * HID);
  cvt_transpose_kernel<<<dim3(HID / 64, HID / 64), 256, 0, stream>>>(
      o_w, ow_t, HID, HID);

  // QKV projection: (4096 x 1024) @ (1024 x 3072) + bias -> bf16
  gemm_bf16_kernel<true><<<dim3(3 * HID / 128, T_TOK / 128), 256, 0, stream>>>(
      hs_b, qkvw_t, qkv_b, qkv, T_TOK, 3 * HID, HID);

  // RoPE + head split (Q pre-scaled, V stored transposed per head)
  rope_split_kernel<<<(T_TOK * HID) / 256, 256, 0, stream>>>(
      qkv, cosp, sinp, qbuf, kbuf, vtbuf);

  // Flash attention over valid tokens (padding keys contribute ~0 and are skipped)
  attn_kernel<<<BATCH * NHEAD * (VALIDT / 64), 128, 0, stream>>>(
      qbuf, kbuf, vtbuf, ctxb);

  // Output projection: (4096 x 1024) @ (1024 x 1024) + bias -> f32 out
  gemm_bf16_kernel<false><<<dim3(HID / 128, T_TOK / 128), 256, 0, stream>>>(
      ctxb, ow_t, o_b, d_out, T_TOK, HID, HID);
}
